// MultiHeadAttention_70420283785785
// MI455X (gfx1250) — compile-verified, matrix operand order checked by reading, not run on hardware
//
#include <hip/hip_runtime.h>
#include <hip/hip_bf16.h>

// ---------------------------------------------------------------------------
// MI455X (gfx1250) multi-head attention forward, bf16 WMMA pipeline.
// wave32, v_wmma_f32_16x16x32_bf16 for all three matrix stages.
// Software-pipelined: double-buffered GLOBAL_LOAD_ASYNC_TO_LDS_B128 staging
// (next tile's copies stay in flight across the WMMA phase, s_wait_asynccnt 4),
// all WMMA operands loaded as contiguous-K row fragments (ds_load_b128 pairs).
// ---------------------------------------------------------------------------

typedef __attribute__((ext_vector_type(16))) __bf16 v16bf;
typedef __attribute__((ext_vector_type(8)))  float  v8f;

#define WMMA_BF16(a, b, c) \
  __builtin_amdgcn_wmma_f32_16x16x32_bf16(false, (a), false, (b), (short)0, (c), false, false)

constexpr int kB  = 4;
constexpr int kS  = 2048;
constexpr int kD  = 1024;
constexpr int kH  = 16;
constexpr int kHd = 64;
constexpr int kM  = kB * kS;     // 8192 total rows
constexpr int kNq = 3 * kD;      // 3072 qkv columns

// ---------------------------------------------------------------------------
// Async global->LDS staging (GLOBAL_LOAD_ASYNC_TO_LDS_B128, ASYNCcnt).
// Signature probe-confirmed in round 2:
//   void __builtin_amdgcn_global_load_async_to_lds_b128(
//       int __vector(4) addrspace(1)* src, int __vector(4) addrspace(3)* dst,
//       int offset, int cpol);
// ---------------------------------------------------------------------------
#if defined(__AMDGCN__) && __has_builtin(__builtin_amdgcn_global_load_async_to_lds_b128)
#define HAS_ASYNC_LDS 1
#define AS1 __attribute__((address_space(1)))
#define AS3 __attribute__((address_space(3)))
typedef int v4i_gl __attribute__((vector_size(16)));   // gcc-style int __vector(4)
__device__ __forceinline__ AS3 v4i_gl* lds_cast(void* p) {
  // Low 32 bits of a generic shared pointer are the LDS byte offset
  // (ISA 10.2 aperture mapping); AS3 pointers are 32-bit.
  return (AS3 v4i_gl*)(unsigned int)(unsigned long long)p;
}
__device__ __forceinline__ AS1 v4i_gl* gl_cast(const void* p) {
  return (AS1 v4i_gl*)(unsigned long long)p;           // also drops const
}
#else
#define HAS_ASYNC_LDS 0
#endif

// 16-byte global -> LDS staging element.
__device__ __forceinline__ void cp16(void* dst, const void* src) {
#if HAS_ASYNC_LDS
  __builtin_amdgcn_global_load_async_to_lds_b128(gl_cast(src), lds_cast(dst), 0, 0);
#else
  *(uint4*)dst = *(const uint4*)src;
#endif
}

// Wait until all of this wave's async copies are complete.
__device__ __forceinline__ void cp_wait_all() {
#if HAS_ASYNC_LDS
#if __has_builtin(__builtin_amdgcn_s_wait_asynccnt)
  __builtin_amdgcn_s_wait_asynccnt(0);
#else
  asm volatile("s_wait_asynccnt 0" ::: "memory");
#endif
#endif
}

// Wait until <= 4 async copies remain (async loads complete in order, so with
// 4 current-tile + 4 next-tile copies in flight this retires the current tile
// while the next tile's copies overlap the compute phase).
__device__ __forceinline__ void cp_wait_tail4() {
#if HAS_ASYNC_LDS
#if __has_builtin(__builtin_amdgcn_s_wait_asynccnt)
  __builtin_amdgcn_s_wait_asynccnt(4);
#else
  asm volatile("s_wait_asynccnt 4" ::: "memory");
#endif
#endif
}

// ---------------------------------------------------------------------------
// Fragment loader (ISA 7.12.2 layouts, wave32).
// A-fragment (16x32 bf16): lane L holds row M = L&15; K-base = (L>>4)*8;
//   f[0..7]  = K(base+0..7), f[8..15] = K(base+16..23).
// B-fragment mirrors it with N = L&15, so a B fragment of B[k][n] is a row
// fragment of the transposed tile Bt[n][k].
// C/D (16x16 f32): VGPR r holds row (r + 8*(L>>4)), col N = L&15.
// ---------------------------------------------------------------------------
__device__ __forceinline__ v16bf frag_ld_rowK(const __bf16* p, int ld, int row0, int k0) {
  const int lane = threadIdx.x & 31;
  const __bf16* q = p + (row0 + (lane & 15)) * ld + k0 + ((lane >> 4) << 3);
  struct { uint4 a, b; } t;
  t.a = *(const uint4*)q;
  t.b = *(const uint4*)(q + 16);
  return __builtin_bit_cast(v16bf, t);
}

// ---------------------------------------------------------------------------
// fp32 -> bf16 conversion (grid-stride), row-major preserved (for x).
// ---------------------------------------------------------------------------
__global__ __launch_bounds__(256) void cvt_f32_bf16(const float* __restrict__ in,
                                                    __bf16* __restrict__ out, int n) {
  int i = blockIdx.x * blockDim.x + threadIdx.x;
  const int stride = gridDim.x * blockDim.x;
  for (; i < n; i += stride) out[i] = (__bf16)in[i];
}

// fp32 [K x N] -> bf16 [N x K] transpose (64x64 LDS tiles, coalesced both ways).
__global__ __launch_bounds__(256) void cvt_transpose_bf16(const float* __restrict__ in,
                                                          __bf16* __restrict__ out,
                                                          int K, int N) {
  __shared__ float tile[64][65];
  const int k0 = blockIdx.y * 64;
  const int n0 = blockIdx.x * 64;
#pragma unroll
  for (int t = 0; t < 16; ++t) {
    const int c = threadIdx.x + t * 256;       // 4096 elements
    const int r = c >> 6, cc = c & 63;
    tile[r][cc] = in[(size_t)(k0 + r) * N + n0 + cc];
  }
  __syncthreads();
#pragma unroll
  for (int t = 0; t < 16; ++t) {
    const int c = threadIdx.x + t * 256;
    const int r = c >> 6, cc = c & 63;         // r: n index, cc: k index
    out[(size_t)(n0 + r) * K + k0 + cc] = (__bf16)tile[cc][r];
  }
}

// ---------------------------------------------------------------------------
// Shared 128x128 GEMM core: 256 threads = 8 waves in a 2(M) x 4(N) grid,
// each wave computes a 64x32 tile = 4x2 WMMA accumulators. BK = 32.
// Double-buffered async staging; B comes from the transposed weights Bt[N][K].
// ---------------------------------------------------------------------------
__device__ __forceinline__ void gemm_core_128x128(
    const __bf16* __restrict__ A, int lda,
    const __bf16* __restrict__ Bt, int ldbt,
    int m0, int n0, int Ktot, v8f acc[4][2]) {
  __shared__ __bf16 As[2][128][40];
  __shared__ __bf16 Bts[2][128][40];
  const int tid = threadIdx.x;
  const int w = tid >> 5;
  const int wm = w >> 2;          // 0..1
  const int wn = w & 3;           // 0..3

  auto stage = [&](int bufi, int kk) {          // 4 async b128 per thread
#pragma unroll
    for (int t = 0; t < 2; ++t) {
      const int c = tid + t * 256;
      const int r = c >> 2, kc = (c & 3) * 8;   // 128 rows x 4 chunks
      cp16(&As[bufi][r][kc],  A  + (size_t)(m0 + r) * lda  + kk + kc);
      cp16(&Bts[bufi][r][kc], Bt + (size_t)(n0 + r) * ldbt + kk + kc);
    }
  };

  stage(0, 0);
  for (int k0 = 0; k0 < Ktot; k0 += 32) {
    const int cur = (k0 >> 5) & 1;
    if (k0 + 32 < Ktot) {
      stage(cur ^ 1, k0 + 32);                  // overlap next tile's copies
      if (k0 + 64 < Ktot)                       // hint tile after that
        __builtin_prefetch(A + (size_t)(m0 + (tid >> 2)) * lda + (k0 + 64) + (tid & 3) * 8, 0, 1);
      cp_wait_tail4();                          // current tile retired
    } else {
      cp_wait_all();
    }
    __syncthreads();

    v16bf a[4], bf[2];
#pragma unroll
    for (int tm = 0; tm < 4; ++tm)
      a[tm] = frag_ld_rowK(&As[cur][0][0],  40, wm * 64 + tm * 16, 0);
#pragma unroll
    for (int tn = 0; tn < 2; ++tn)
      bf[tn] = frag_ld_rowK(&Bts[cur][0][0], 40, wn * 32 + tn * 16, 0);
#pragma unroll
    for (int tm = 0; tm < 4; ++tm)
#pragma unroll
      for (int tn = 0; tn < 2; ++tn)
        acc[tm][tn] = WMMA_BF16(a[tm], bf[tn], acc[tm][tn]);
    __syncthreads();                            // buffer reuse fence
  }
}

// ---------------------------------------------------------------------------
// QKV GEMM: [8192 x 1024] @ Wt[3072 x 1024] + bias. Q,K scatter to head-major
// [B,H,S,Hd]; V scatters TRANSPOSED to [B,H,Hd,S] so attention's P@V B-matrix
// fragments are contiguous-K loads.
// ---------------------------------------------------------------------------
__global__ __launch_bounds__(256) void qkv_gemm(
    const __bf16* __restrict__ X, const __bf16* __restrict__ Wt,
    const float* __restrict__ bias,
    __bf16* __restrict__ Qo, __bf16* __restrict__ Ko, __bf16* __restrict__ Vo) {
  const int n0 = blockIdx.x * 128;
  const int m0 = blockIdx.y * 128;
  const int lane = threadIdx.x & 31;
  const int w = threadIdx.x >> 5, wm = w >> 2, wn = w & 3;

  v8f acc[4][2] = {};
  gemm_core_128x128(X, kD, Wt, kD, m0, n0, kD, acc);

#pragma unroll
  for (int tm = 0; tm < 4; ++tm)
#pragma unroll
    for (int tn = 0; tn < 2; ++tn) {
      const int n = n0 + wn * 32 + tn * 16 + (lane & 15);
      const int mb = m0 + wm * 64 + tm * 16 + ((lane >> 4) << 3);
      const float bv = bias[n];
      const int sel = n >> 10;                  // 0:q 1:k 2:v
      const int rem = n & 1023;
      const int h = rem >> 6, hd = rem & 63;
#pragma unroll
      for (int r = 0; r < 8; ++r) {
        const int m = mb + r;
        const int b = m >> 11, s = m & 2047;
        const __bf16 val = (__bf16)(acc[tm][tn][r] + bv);
        if (sel == 0)
          Qo[(size_t)(((b << 4) + h) * kS + s) * kHd + hd] = val;
        else if (sel == 1)
          Ko[(size_t)(((b << 4) + h) * kS + s) * kHd + hd] = val;
        else
          Vo[(size_t)(((b << 4) + h) * kHd + hd) * kS + s] = val;   // transposed
      }
    }
}

// ---------------------------------------------------------------------------
// Flash attention: one (b,h) x 128 query rows per workgroup; 8 waves each own
// a 16-row strip. Q fragments are hoisted into registers (staged once through
// the K/V buffer); K/V tiles are double-buffered async copies. Online softmax
// in fp32; P re-laid-out via LDS to A-fragments. V arrives transposed [Hd][S].
// ---------------------------------------------------------------------------
__global__ __launch_bounds__(256) void attn_kernel(
    const __bf16* __restrict__ Q, const __bf16* __restrict__ K,
    const __bf16* __restrict__ Vt, __bf16* __restrict__ O) {
  __shared__ __bf16 KV[2][2][64][72];           // [stage][K=0 / Vt=1][row][col]
  __shared__ __bf16 Ps[8][16][72];

  const int bh = blockIdx.y;                    // 0..63
  const int q0 = blockIdx.x * 128;
  const int tid = threadIdx.x, lane = tid & 31, w = tid >> 5;
  const int half = lane >> 4;

  const __bf16* Qg  = Q  + (size_t)bh * kS * kHd;  // [S][Hd]
  const __bf16* Kg  = K  + (size_t)bh * kS * kHd;  // [S][Hd]
  const __bf16* Vgt = Vt + (size_t)bh * kHd * kS;  // [Hd][S]

  // Stage the 128x64 query block once through KV[0] (contiguous 128 rows),
  // hoist this wave's two A-fragments into registers, then recycle the buffer.
  __bf16 (*Qstage)[72] = &KV[0][0][0];
#pragma unroll
  for (int t = 0; t < 4; ++t) {
    const int c = tid + t * 256;                // 1024 chunks of 8 bf16
    const int r = c >> 3, kc = (c & 7) * 8;
    cp16(&Qstage[r][kc], Qg + (size_t)(q0 + r) * kHd + kc);
  }
  cp_wait_all();
  __syncthreads();
  const v16bf aq0 = frag_ld_rowK(&Qstage[0][0], 72, w * 16, 0);
  const v16bf aq1 = frag_ld_rowK(&Qstage[0][0], 72, w * 16, 32);
  __syncthreads();                              // Q consumed; buffer reusable

  auto stage_kv = [&](int bufi, int kb) {       // 4 async b128 per thread
#pragma unroll
    for (int t = 0; t < 2; ++t) {
      const int c = tid + t * 256;              // 512 chunks of 8 bf16 each
      const int r = c >> 3, kc = (c & 7) * 8;
      cp16(&KV[bufi][0][r][kc], Kg  + (size_t)(kb + r) * kHd + kc);  // keys
      cp16(&KV[bufi][1][r][kc], Vgt + (size_t)r * kS + kb + kc);     // hd rows
    }
  };

  v8f o[4] = {};
  float mrow[8], lrow[8];
#pragma unroll
  for (int r = 0; r < 8; ++r) { mrow[r] = -1e30f; lrow[r] = 0.f; }

  stage_kv(0, 0);
  for (int kb = 0; kb < kS; kb += 64) {
    const int cur = (kb >> 6) & 1;
    if (kb + 64 < kS) {
      stage_kv(cur ^ 1, kb + 64);               // overlap next block's copies
      cp_wait_tail4();
    } else {
      cp_wait_all();
    }
    __syncthreads();
    const __bf16* Kst = &KV[cur][0][0][0];
    const __bf16* Vst = &KV[cur][1][0][0];

    // S = Q @ K^T for this wave's 16-row strip: 4 column tiles over 64 keys.
    v8f s[4];
    {
      const v8f z = {0.f, 0.f, 0.f, 0.f, 0.f, 0.f, 0.f, 0.f};
#pragma unroll
      for (int t = 0; t < 4; ++t) s[t] = z;
    }
#pragma unroll
    for (int t = 0; t < 4; ++t) {
      const v16bf bk0 = frag_ld_rowK(Kst, 72, t * 16, 0);   // K^T access
      s[t] = WMMA_BF16(aq0, bk0, s[t]);
      const v16bf bk1 = frag_ld_rowK(Kst, 72, t * 16, 32);
      s[t] = WMMA_BF16(aq1, bk1, s[t]);
    }
#pragma unroll
    for (int t = 0; t < 4; ++t)
#pragma unroll
      for (int r = 0; r < 8; ++r) s[t][r] *= 0.125f;  // 1/sqrt(Hd)

    // Online softmax (row r lives in VGPR r across the lane-half's 16 lanes).
#pragma unroll
    for (int r = 0; r < 8; ++r) {
      float rm = fmaxf(fmaxf(s[0][r], s[1][r]), fmaxf(s[2][r], s[3][r]));
      rm = fmaxf(rm, __shfl_xor(rm, 1, 32));
      rm = fmaxf(rm, __shfl_xor(rm, 2, 32));
      rm = fmaxf(rm, __shfl_xor(rm, 4, 32));
      rm = fmaxf(rm, __shfl_xor(rm, 8, 32));
      const float mnew = fmaxf(mrow[r], rm);
      const float alpha = __expf(mrow[r] - mnew);
      float rs = 0.f;
#pragma unroll
      for (int t = 0; t < 4; ++t) {
        const float p = __expf(s[t][r] - mnew);
        s[t][r] = p;
        rs += p;
      }
      rs += __shfl_xor(rs, 1, 32);
      rs += __shfl_xor(rs, 2, 32);
      rs += __shfl_xor(rs, 4, 32);
      rs += __shfl_xor(rs, 8, 32);
      lrow[r] = lrow[r] * alpha + rs;
      mrow[r] = mnew;
#pragma unroll
      for (int t = 0; t < 4; ++t) o[t][r] *= alpha;
    }

    // Re-layout P (C-layout fp32) -> LDS bf16 -> A fragments. Per-wave region;
    // LDS is in-order within a wave, the wait is belt-and-suspenders.
#pragma unroll
    for (int t = 0; t < 4; ++t)
#pragma unroll
      for (int r = 0; r < 8; ++r)
        Ps[w][r + (half << 3)][t * 16 + (lane & 15)] = (__bf16)s[t][r];
    asm volatile("s_wait_dscnt 0" ::: "memory");

    // O += P @ V: B[k][n] = V[key=k][hd=n] = Vt[n][k] -> row fragments of Vst.
#pragma unroll
    for (int ks = 0; ks < 2; ++ks) {
      const v16bf ap = frag_ld_rowK(&Ps[w][0][0], 72, 0, ks * 32);
#pragma unroll
      for (int t = 0; t < 4; ++t) {
        const v16bf bv = frag_ld_rowK(Vst, 72, t * 16, ks * 32);
        o[t] = WMMA_BF16(ap, bv, o[t]);
      }
    }
    __syncthreads();                            // buffer reuse fence
  }

  // Normalize and store to [B,S,H,Hd] (= [8192 x 1024] row-major) bf16.
  const int b = bh >> 4, h = bh & 15;
#pragma unroll
  for (int r = 0; r < 8; ++r) {
    const float inv = 1.f / lrow[r];
    const int srow = q0 + w * 16 + (half << 3) + r;
    const size_t base = (size_t)(b * kS + srow) * kD + h * kHd;
#pragma unroll
    for (int t = 0; t < 4; ++t)
      O[base + t * 16 + (lane & 15)] = (__bf16)(o[t][r] * inv);
  }
}

// ---------------------------------------------------------------------------
// Output projection: [8192 x 1024] @ Wt[1024 x 1024] + bias -> fp32 d_out.
// ---------------------------------------------------------------------------
__global__ __launch_bounds__(256) void proj_gemm(
    const __bf16* __restrict__ A, const __bf16* __restrict__ Wt,
    const float* __restrict__ bias, float* __restrict__ out) {
  const int n0 = blockIdx.x * 128;
  const int m0 = blockIdx.y * 128;
  const int lane = threadIdx.x & 31;
  const int w = threadIdx.x >> 5, wm = w >> 2, wn = w & 3;

  v8f acc[4][2] = {};
  gemm_core_128x128(A, kD, Wt, kD, m0, n0, kD, acc);

#pragma unroll
  for (int tm = 0; tm < 4; ++tm)
#pragma unroll
    for (int tn = 0; tn < 2; ++tn) {
      const int n = n0 + wn * 32 + tn * 16 + (lane & 15);
      const int mb = m0 + wm * 64 + tm * 16 + ((lane >> 4) << 3);
      const float bv = bias[n];
#pragma unroll
      for (int r = 0; r < 8; ++r)
        out[(size_t)(mb + r) * kD + n] = acc[tm][tn][r] + bv;
    }
}

// ---------------------------------------------------------------------------
// Host-side launcher
// ---------------------------------------------------------------------------
extern "C" void kernel_launch(void* const* d_in, const int* in_sizes, int n_in,
                              void* d_out, int out_size, void* d_ws, size_t ws_size,
                              hipStream_t stream) {
  (void)in_sizes; (void)n_in; (void)out_size; (void)ws_size;
  const float* x      = (const float*)d_in[0];
  const float* w_qkv  = (const float*)d_in[1];
  const float* b_qkv  = (const float*)d_in[2];
  const float* w_proj = (const float*)d_in[3];
  const float* b_proj = (const float*)d_in[4];
  float* out = (float*)d_out;

  // bf16 workspace layout (all 16B aligned): ~88 MB total.
  __bf16* xb     = (__bf16*)d_ws;                       // [8192][1024]
  __bf16* wqkvt  = xb    + (size_t)kM * kD;             // [3072][1024] (transposed)
  __bf16* wprojt = wqkvt + (size_t)kNq * kD;            // [1024][1024] (transposed)
  __bf16* qbuf   = wprojt + (size_t)kD * kD;            // [B,H,S,Hd]
  __bf16* kbuf   = qbuf  + (size_t)kB * kH * kS * kHd;  // [B,H,S,Hd]
  __bf16* vbuf   = kbuf  + (size_t)kB * kH * kS * kHd;  // [B,H,Hd,S] (transposed)
  __bf16* aob    = vbuf  + (size_t)kB * kH * kS * kHd;  // [B,S,D]

  cvt_f32_bf16<<<2048, 256, 0, stream>>>(x, xb, kM * kD);
  cvt_transpose_bf16<<<dim3(kNq / 64, kD / 64), 256, 0, stream>>>(w_qkv, wqkvt, kD, kNq);
  cvt_transpose_bf16<<<dim3(kD / 64, kD / 64), 256, 0, stream>>>(w_proj, wprojt, kD, kD);

  qkv_gemm<<<dim3(kNq / 128, kM / 128), 256, 0, stream>>>(xb, wqkvt, b_qkv,
                                                          qbuf, kbuf, vbuf);
  attn_kernel<<<dim3(kS / 128, kB * kH), 256, 0, stream>>>(qbuf, kbuf, vbuf, aob);
  proj_gemm<<<dim3(kD / 128, kM / 128), 256, 0, stream>>>(aob, wprojt, b_proj, out);
}
